// VDP_first_Conv_27745488732891
// MI455X (gfx1250) — compile-verified, matrix operand order checked by reading, not run on hardware
//
#include <hip/hip_runtime.h>
#include <hip/hip_bf16.h>
#include <math.h>

typedef __attribute__((ext_vector_type(16))) _Float16 v16h;
typedef __attribute__((ext_vector_type(8)))  float    v8f;

#define H_IN    224
#define W_IN    224
#define CIN     3
#define COUT    64
#define HO      220
#define WO      220
#define KPATCH  75          // 5*5*3
#define KPAD    96          // 3 WMMA K-steps of 32
#define TILE_W  64          // output columns per workgroup
#define APITCH  104         // padded LDS row pitch in halves (208B: conflict-free b128 lanes)

__device__ __forceinline__ float softplus_f(float x) {
    // numerically stable log(1+exp(x))
    return fmaxf(x, 0.0f) + log1pf(expf(-fabsf(x)));
}

union Pk8 { _Float16 h[8]; uint4 q; };          // 8 halves == one b128 store
union FragAB { v16h v; uint4 q[2]; };           // WMMA A/B fragment == two b128 loads

__global__ __launch_bounds__(128) void vdp_conv_kernel(
    const float* __restrict__ mu_in,    // [32,224,224,3]
    const float* __restrict__ w_mu,     // [5,5,3,64] == [75,64] row-major
    const float* __restrict__ w_sigma,  // [64]
    float* __restrict__ mu_out,         // [32,220,220,64]
    float* __restrict__ sigma_out)      // [32,220,220,64]
{
    // K-contiguous im2col panels, f32 split into f16 hi + f16 lo (3-product fp32 emulation)
    __shared__ _Float16 a_hi[TILE_W * APITCH];
    __shared__ _Float16 a_lo[TILE_W * APITCH];
    __shared__ _Float16 b_hi[COUT * APITCH];
    __shared__ _Float16 b_lo[COUT * APITCH];
    __shared__ float    sq_lds[TILE_W];
    __shared__ float    spws[COUT];

    const int tid = threadIdx.x;
    const int bid = blockIdx.x;
    const int wt  = bid & 3;               // 4 column tiles of 64 (last partial: 28)
    const int rem = bid >> 2;
    const int ho  = rem % HO;
    const int b   = rem / HO;
    const int wo0 = wt * TILE_W;

    if (tid < TILE_W) sq_lds[tid] = 0.0f;
    if (tid < COUT)   spws[tid]   = softplus_f(w_sigma[tid]);
    __syncthreads();

    // ---- Stage A panel: task = (pos, K-quarter of 24); b128 stores, hi/lo separate ----
    for (int task = tid; task < TILE_W * 4; task += 128) {
        const int pos = task >> 2;
        const int q   = task & 3;
        float x2s = 0.0f;
        #pragma unroll
        for (int blk = 0; blk < 3; ++blk) {
            Pk8 phi, plo;
            #pragma unroll
            for (int j = 0; j < 8; ++j) {
                int k = q * 24 + blk * 8 + j;
                float x = 0.0f;
                if (k < KPATCH) {
                    int kh = k / 15, r = k % 15;
                    int kw = r / 3,  c = r % 3;
                    int gcol = wo0 + pos + kw;
                    if (gcol > W_IN - 1) gcol = W_IN - 1;   // clamp; OOB positions never stored
                    x = mu_in[(((size_t)b * H_IN + (ho + kh)) * W_IN + gcol) * CIN + c];
                }
                _Float16 hi = (_Float16)x;
                phi.h[j] = hi;
                plo.h[j] = (_Float16)(x - (float)hi);
                x2s += x * x;
            }
            const int off = pos * APITCH + q * 24 + blk * 8;   // byte offset multiple of 16
            *(uint4*)&a_hi[off] = phi.q;
            *(uint4*)&a_lo[off] = plo.q;
        }
        atomicAdd(&sq_lds[pos], x2s);      // ds_add_f32: variance box-sum for free
    }

    // ---- Stage B panel: task = (n, K-quarter); transpose w_mu[k][n] -> b[n][k] ----
    for (int task = tid; task < COUT * 4; task += 128) {
        const int n = task >> 2;
        const int q = task & 3;
        #pragma unroll
        for (int blk = 0; blk < 3; ++blk) {
            Pk8 phi, plo;
            #pragma unroll
            for (int j = 0; j < 8; ++j) {
                int k = q * 24 + blk * 8 + j;
                float w = (k < KPATCH) ? w_mu[k * COUT + n] : 0.0f;
                _Float16 hi = (_Float16)w;
                phi.h[j] = hi;
                plo.h[j] = (_Float16)(w - (float)hi);
            }
            const int off = n * APITCH + q * 24 + blk * 8;
            *(uint4*)&b_hi[off] = phi.q;
            *(uint4*)&b_lo[off] = plo.q;
        }
    }
    __syncthreads();

    // ---- Implicit GEMM: wave w -> N-tile w (16 channels), 4 M-tiles of 16 positions ----
    const int wave    = tid >> 5;
    const int lane    = tid & 31;
    const int nloc    = lane & 15;
    const int halfsel = lane >> 4;          // 0: lanes 0-15, 1: lanes 16-31
    const int klbase  = halfsel * 8;        // per-lane K base within a 32-wide K step
    const int nglob   = wave * 16 + nloc;

    const _Float16* bh_row = &b_hi[nglob * APITCH];
    const _Float16* bl_row = &b_lo[nglob * APITCH];

    v8f acc[4] = {};

    #pragma unroll
    for (int ks = 0; ks < 3; ++ks) {
        const int koff = ks * 32 + klbase;

        FragAB fbh, fbl;
        fbh.q[0] = *(const uint4*)(bh_row + koff);
        fbh.q[1] = *(const uint4*)(bh_row + koff + 16);
        fbl.q[0] = *(const uint4*)(bl_row + koff);
        fbl.q[1] = *(const uint4*)(bl_row + koff + 16);

        #pragma unroll
        for (int m = 0; m < 4; ++m) {
            const int pos = m * 16 + nloc;
            FragAB fah, fal;
            const _Float16* ah_row = &a_hi[pos * APITCH];
            const _Float16* al_row = &a_lo[pos * APITCH];
            fah.q[0] = *(const uint4*)(ah_row + koff);
            fah.q[1] = *(const uint4*)(ah_row + koff + 16);
            fal.q[0] = *(const uint4*)(al_row + koff);
            fal.q[1] = *(const uint4*)(al_row + koff + 16);

            // fp32-accurate split product: ah*bh + al*bh + ah*bl
            acc[m] = __builtin_amdgcn_wmma_f32_16x16x32_f16(
                false, fah.v, false, fbh.v, (short)0, acc[m], false, false);
            acc[m] = __builtin_amdgcn_wmma_f32_16x16x32_f16(
                false, fal.v, false, fbh.v, (short)0, acc[m], false, false);
            acc[m] = __builtin_amdgcn_wmma_f32_16x16x32_f16(
                false, fah.v, false, fbl.v, (short)0, acc[m], false, false);
        }
    }

    // ---- Store mu_out: lane L holds (M = r + 8*halfsel + 16*m, N = nglob) ----
    const size_t rowbase = (((size_t)b * HO + ho) * WO) * COUT;
    #pragma unroll
    for (int m = 0; m < 4; ++m) {
        #pragma unroll
        for (int r = 0; r < 8; ++r) {
            int M  = m * 16 + r + halfsel * 8;
            int wo = wo0 + M;
            if (wo < WO)
                mu_out[rowbase + (size_t)wo * COUT + nglob] = acc[m][r];
        }
    }

    // ---- Store Sigma_out = softplus(softplus(w_sigma)[n] * sq[pos] / 75) ----
    for (int e = tid; e < TILE_W * COUT; e += 128) {
        int pos = e >> 6;
        int n   = e & 63;
        int wo  = wo0 + pos;
        if (wo < WO) {
            float z = spws[n] * (sq_lds[pos] * (1.0f / (float)KPATCH));
            sigma_out[rowbase + (size_t)wo * COUT + n] = softplus_f(z);
        }
    }
}

// ---- KL scalar: 0.5*mean( log(Pv)-1-ws + softplus(ws)/Pv + w_mu^2/Pv ) ----
__global__ __launch_bounds__(256) void vdp_kl_kernel(
    const float* __restrict__ w_mu, const float* __restrict__ w_sigma,
    float* __restrict__ out_kl)
{
    __shared__ float red[256];
    const int tid = threadIdx.x;
    float s = 0.0f;
    for (int i = tid; i < KPATCH * COUT; i += 256) {
        float w = w_mu[i];
        s += w * w;
    }
    s *= 100.0f;                               // / PRIOR_VAR
    if (tid < COUT) {
        float ws = w_sigma[tid];
        s += 75.0f * (-ws + softplus_f(ws) * 100.0f);
    }
    red[tid] = s;
    __syncthreads();
    for (int o = 128; o > 0; o >>= 1) {
        if (tid < o) red[tid] += red[tid + o];
        __syncthreads();
    }
    if (tid == 0) {
        float mean = red[0] / (float)(KPATCH * COUT) + (logf(0.01f) - 1.0f);
        out_kl[0] = 0.5f * mean;
    }
}

extern "C" void kernel_launch(void* const* d_in, const int* in_sizes, int n_in,
                              void* d_out, int out_size, void* d_ws, size_t ws_size,
                              hipStream_t stream) {
    (void)in_sizes; (void)n_in; (void)out_size; (void)d_ws; (void)ws_size;
    const float* mu_in   = (const float*)d_in[0];
    const float* w_mu    = (const float*)d_in[1];
    const float* w_sigma = (const float*)d_in[2];

    const size_t n_out_elems = (size_t)32 * HO * WO * COUT;   // 99,123,200
    float* mu_out    = (float*)d_out;
    float* sigma_out = mu_out + n_out_elems;
    float* kl        = mu_out + 2 * n_out_elems;

    dim3 grid(32 * HO * 4);   // (batch, output row, 64-col tile)
    vdp_conv_kernel<<<grid, 128, 0, stream>>>(mu_in, w_mu, w_sigma, mu_out, sigma_out);
    vdp_kl_kernel<<<1, 256, 0, stream>>>(w_mu, w_sigma, kl);
}
